// GAT_3_9706626090120
// MI455X (gfx1250) — matrix operand order repaired and checked by reading, not compiled, not run on hardware
//
#include <hip/hip_runtime.h>

// ---------- CDNA5 WMMA types ----------
typedef __attribute__((ext_vector_type(16))) __bf16 v16bf;
typedef __attribute__((ext_vector_type(8)))  float  v8f;
typedef __attribute__((ext_vector_type(4)))  int    v4i;

typedef __attribute__((address_space(1))) v4i* as1_v4i_ptr;  // global
typedef __attribute__((address_space(3))) v4i* as3_v4i_ptr;  // LDS

// Native f32 -> bf16 (backend picks gfx1250 HW cvt)
__device__ __forceinline__ __bf16 cvt_bf16(float f) { return (__bf16)f; }

// Async global->LDS path (gfx1250), guarded so absence degrades gracefully
#if defined(__AMDGCN__) && __has_builtin(__builtin_amdgcn_global_load_async_to_lds_b128) && \
    __has_builtin(__builtin_amdgcn_s_wait_asynccnt)
#define USE_ASYNC_LDS 1
#else
#define USE_ASYNC_LDS 0
#endif

// ============================================================
// GEMM: Hout[M x 64] = A[M x K] * W[K x 64]   (K = 128 or 64, compile-time)
// block = 128 threads = 4 waves; wave w owns N-tile cols [16w, 16w+16)
// W staged in LDS (async global->LDS copy); B-fragments resident in VGPRs.
// Fully unrolled K chain of v_wmma_f32_16x16x32_bf16, fp32 accumulate.
// ============================================================
template <int K>
__global__ __launch_bounds__(128)
void gat_gemm_wmma(const float* __restrict__ A, const float* __restrict__ W,
                   float* __restrict__ Hout, int M) {
  constexpr int KS = K / 32;
  __shared__ float ldsWf[K * 64];            // max 32 KB of 320 KB WGP LDS

  const int tid = threadIdx.x;
#if USE_ASYNC_LDS
  {
    for (int i = tid; i < (K * 64) / 4; i += 128) {
      v4i* gsrc = (v4i*)const_cast<float*>(W + i * 4);
      v4i* ldst = (v4i*)(&ldsWf[i * 4]);
      __builtin_amdgcn_global_load_async_to_lds_b128(
          (as1_v4i_ptr)gsrc, (as3_v4i_ptr)ldst, 0, 0);
    }
    __builtin_amdgcn_s_wait_asynccnt(0);
  }
#else
  for (int i = tid; i < K * 64; i += 128) ldsWf[i] = W[i];
#endif
  __syncthreads();

  const int lane = tid & 31;
  const int col  = lane & 15;     // N index within tile / M row index for A frag
  const int half = lane >> 4;     // 0: lanes 0-15, 1: lanes 16-31
  const int col0 = (tid >> 5) * 16;

  // B fragments: 32x16 bf16 layout -> lanes 0-15: K=k0..k0+15, lanes 16-31: K=k0+16..k0+31
  v16bf bfrag[KS];
#pragma unroll
  for (int ks = 0; ks < KS; ++ks) {
    const int kb = ks * 32 + half * 16;
    v16bf b;
#pragma unroll
    for (int t = 0; t < 16; ++t) b[t] = cvt_bf16(ldsWf[(kb + t) * 64 + (col0 + col)]);
    bfrag[ks] = b;
  }

  const int mTiles = (M + 15) >> 4;
  for (int mt = blockIdx.x; mt < mTiles; mt += gridDim.x) {
    const int row  = mt * 16 + col;
    const int rowc = row < M ? row : (M - 1);
    const float* arow = A + (size_t)rowc * K;

    v8f c = {};
#pragma unroll
    for (int ks = 0; ks < KS; ++ks) {
      // A fragment 16x32 bf16 layout: lane holds row (lane&15),
      // K = ks*32 + half*16 + {0..15}
      const int kb = ks * 32 + half * 16;
      const float4 x0 = *reinterpret_cast<const float4*>(arow + kb);
      const float4 x1 = *reinterpret_cast<const float4*>(arow + kb + 4);
      const float4 x2 = *reinterpret_cast<const float4*>(arow + kb + 8);
      const float4 x3 = *reinterpret_cast<const float4*>(arow + kb + 12);
      v16bf a;
      a[0]  = cvt_bf16(x0.x); a[1]  = cvt_bf16(x0.y); a[2]  = cvt_bf16(x0.z); a[3]  = cvt_bf16(x0.w);
      a[4]  = cvt_bf16(x1.x); a[5]  = cvt_bf16(x1.y); a[6]  = cvt_bf16(x1.z); a[7]  = cvt_bf16(x1.w);
      a[8]  = cvt_bf16(x2.x); a[9]  = cvt_bf16(x2.y); a[10] = cvt_bf16(x2.z); a[11] = cvt_bf16(x2.w);
      a[12] = cvt_bf16(x3.x); a[13] = cvt_bf16(x3.y); a[14] = cvt_bf16(x3.z); a[15] = cvt_bf16(x3.w);
      c = __builtin_amdgcn_wmma_f32_16x16x32_bf16(
          /*neg_a=*/false, a, /*neg_b=*/false, bfrag[ks],
          /*c_mod=*/(short)0, c, /*reuse_a=*/false, /*reuse_b=*/false);
    }

    // C layout: VGPR r holds M = mt*16 + half*8 + r, N = col0+col
    if (mt * 16 + 16 <= M) {
      float* outp = Hout + (size_t)(mt * 16 + half * 8) * 64 + col0 + col;
#pragma unroll
      for (int r = 0; r < 8; ++r) outp[(size_t)r * 64] = c[r];
    } else {
#pragma unroll
      for (int r = 0; r < 8; ++r) {
        const int m = mt * 16 + half * 8 + r;
        if (m < M) Hout[(size_t)m * 64 + col0 + col] = c[r];
      }
    }
  }
}

// ============================================================
// alpha_src[n] = dot(H[n], a_src); alpha_dst[n] = dot(H[n], a_dst)
// one wave per node, coalesced loads, shuffle reduction (wave32)
// ============================================================
__global__ __launch_bounds__(256)
void gat_alpha(const float* __restrict__ H, const float* __restrict__ a_src,
               const float* __restrict__ a_dst, float* __restrict__ as_,
               float* __restrict__ ad_, int N) {
  const int node = blockIdx.x * 8 + (threadIdx.x >> 5);
  if (node >= N) return;
  const int lane = threadIdx.x & 31;
  const float* hrow = H + (size_t)node * 64;
  const float h0 = hrow[lane], h1 = hrow[lane + 32];
  float ps = h0 * a_src[lane] + h1 * a_src[lane + 32];
  float pd = h0 * a_dst[lane] + h1 * a_dst[lane + 32];
#pragma unroll
  for (int off = 16; off > 0; off >>= 1) {
    ps += __shfl_down(ps, off, 32);
    pd += __shfl_down(pd, off, 32);
  }
  if (lane == 0) { as_[node] = ps; ad_[node] = pd; }
}

// ============================================================
// init: accum = 0, m = -inf (bits), denom = 0
// ============================================================
__global__ void gat_init(float* __restrict__ accum, int* __restrict__ mbits,
                         float* __restrict__ denom, int N) {
  const long long stride = (long long)gridDim.x * blockDim.x;
  const long long total = (long long)N * 64;
  for (long long i = (long long)blockIdx.x * blockDim.x + threadIdx.x;
       i < total; i += stride) {
    accum[i] = 0.0f;
    if (i < N) { mbits[i] = (int)0xFF800000; denom[i] = 0.0f; }  // -inf
  }
}

__device__ __forceinline__ float lrelu02(float v) {
  return v > 0.0f ? v : v * 0.2f;
}

// ============================================================
// edge pass 1: segment-max of leaky_relu(alpha_src[s]+alpha_dst[d]) into m[d]
// float atomic-max via signed-max / unsigned-min trick
// ============================================================
__global__ void gat_edge_max(const long long* __restrict__ ei,
                             const float* __restrict__ as_, const float* __restrict__ ad_,
                             int* __restrict__ mbits, long long E) {
  const long long stride = (long long)gridDim.x * blockDim.x;
  for (long long e = (long long)blockIdx.x * blockDim.x + threadIdx.x;
       e < E; e += stride) {
    const int s = (int)ei[e];
    const int d = (int)ei[E + e];
    const float v = lrelu02(as_[s] + ad_[d]);
    if (v >= 0.0f) atomicMax(&mbits[d], __float_as_int(v));
    else           atomicMin((unsigned int*)&mbits[d], __float_as_uint(v));
  }
}

// ============================================================
// edge pass 2: ee = exp(e - m[d]); denom[d] += ee
// ============================================================
__global__ void gat_edge_exp(const long long* __restrict__ ei,
                             const float* __restrict__ as_, const float* __restrict__ ad_,
                             const int* __restrict__ mbits, float* __restrict__ ee,
                             float* __restrict__ denom, long long E) {
  const long long stride = (long long)gridDim.x * blockDim.x;
  for (long long e = (long long)blockIdx.x * blockDim.x + threadIdx.x;
       e < E; e += stride) {
    const int s = (int)ei[e];
    const int d = (int)ei[E + e];
    const float v = lrelu02(as_[s] + ad_[d]);
    const float m = __int_as_float(mbits[d]);
    const float x = __expf(v - m);
    ee[e] = x;
    atomicAdd(&denom[d], x);
  }
}

// ============================================================
// edge pass 3: accum[d,:] += (ee/denom[d]) * H[s,:]
// 64 threads per edge -> coalesced gather + coalesced atomic scatter
// ============================================================
__global__ __launch_bounds__(256)
void gat_edge_agg(const long long* __restrict__ ei, const float* __restrict__ ee,
                  const float* __restrict__ denom, const float* __restrict__ H,
                  float* __restrict__ accum, long long E) {
  const long long idx = (long long)blockIdx.x * blockDim.x + threadIdx.x;
  const long long e = idx >> 6;
  if (e >= E) return;
  const int f = (int)(idx & 63);
  const int s = (int)ei[e];
  const int d = (int)ei[E + e];
  float den = denom[d];
  den = den > 1e-16f ? den : 1e-16f;
  const float w = ee[e] / den;
  atomicAdd(&accum[(size_t)d * 64 + f], w * H[(size_t)s * 64 + f]);
}

// ============================================================
// finalize: out = (accum + bias) [optionally relu]
// ============================================================
__global__ void gat_finalize(const float* __restrict__ accum, const float* __restrict__ bias,
                             float* __restrict__ out, int N, int doRelu) {
  const long long stride = (long long)gridDim.x * blockDim.x;
  const long long total = (long long)N * 64;
  for (long long i = (long long)blockIdx.x * blockDim.x + threadIdx.x;
       i < total; i += stride) {
    float v = accum[i] + bias[(int)(i & 63)];
    if (doRelu) v = v > 0.0f ? v : 0.0f;
    out[i] = v;
  }
}

// ============================================================
// Host orchestration
// ============================================================
extern "C" void kernel_launch(void* const* d_in, const int* in_sizes, int n_in,
                              void* d_out, int out_size, void* d_ws, size_t ws_size,
                              hipStream_t stream) {
  const float*     x   = (const float*)d_in[0];
  const long long* ei  = (const long long*)d_in[1];
  const float*     W1  = (const float*)d_in[2];
  const float*     a1s = (const float*)d_in[3];
  const float*     a1d = (const float*)d_in[4];
  const float*     b1  = (const float*)d_in[5];
  const float*     W2  = (const float*)d_in[6];
  const float*     a2s = (const float*)d_in[7];
  const float*     a2d = (const float*)d_in[8];
  const float*     b2  = (const float*)d_in[9];

  const int Fin = in_sizes[2] / 64;          // W1 is [Fin, 64] -> 128
  const int N   = in_sizes[0] / Fin;         // 100000
  const long long E = in_sizes[1] / 2;       // 1,600,000

  // workspace carve-out
  char* ws = (char*)d_ws;
  size_t off = 0;
  auto carve = [&](size_t bytes) { char* p = ws + off; off = (off + bytes + 255) & ~(size_t)255; return p; };
  float* h     = (float*)carve((size_t)N * 64 * sizeof(float));
  float* out1  = (float*)carve((size_t)N * 64 * sizeof(float));
  float* as_   = (float*)carve((size_t)N * sizeof(float));
  float* ad_   = (float*)carve((size_t)N * sizeof(float));
  int*   mb    = (int*)  carve((size_t)N * sizeof(int));
  float* den   = (float*)carve((size_t)N * sizeof(float));
  float* ee    = (float*)carve((size_t)E * sizeof(float));
  (void)ws_size; (void)n_in; (void)out_size;

  const int mTiles   = (N + 15) / 16;
  const int gemmGrid = mTiles < 2048 ? mTiles : 2048;
  const int alphaGrid = (N + 7) / 8;
  const int edgeGrid  = (int)((E + 255) / 256) < 8192 ? (int)((E + 255) / 256) : 8192;
  const int aggGrid   = (int)((E * 64 + 255) / 256);

  float* outF = (float*)d_out;

  // ---------------- layer 1 ----------------
  if (Fin == 128) gat_gemm_wmma<128><<<gemmGrid, 128, 0, stream>>>(x, W1, h, N);
  else            gat_gemm_wmma<64> <<<gemmGrid, 128, 0, stream>>>(x, W1, h, N);
  gat_alpha<<<alphaGrid, 256, 0, stream>>>(h, a1s, a1d, as_, ad_, N);
  gat_init<<<4096, 256, 0, stream>>>(out1, mb, den, N);
  gat_edge_max<<<edgeGrid, 256, 0, stream>>>(ei, as_, ad_, mb, E);
  gat_edge_exp<<<edgeGrid, 256, 0, stream>>>(ei, as_, ad_, mb, ee, den, E);
  gat_edge_agg<<<aggGrid, 256, 0, stream>>>(ei, ee, den, h, out1, E);
  gat_finalize<<<4096, 256, 0, stream>>>(out1, b1, out1, N, /*relu=*/1);

  // ---------------- layer 2 ----------------
  gat_gemm_wmma<64><<<gemmGrid, 128, 0, stream>>>(out1, W2, h, N);
  gat_alpha<<<alphaGrid, 256, 0, stream>>>(h, a2s, a2d, as_, ad_, N);
  gat_init<<<4096, 256, 0, stream>>>(outF, mb, den, N);
  gat_edge_max<<<edgeGrid, 256, 0, stream>>>(ei, as_, ad_, mb, E);
  gat_edge_exp<<<edgeGrid, 256, 0, stream>>>(ei, as_, ad_, mb, ee, den, E);
  gat_edge_agg<<<aggGrid, 256, 0, stream>>>(ei, ee, den, h, outF, E);
  gat_finalize<<<4096, 256, 0, stream>>>(outF, b2, outF, N, /*relu=*/0);
}
